// SelfAttention_2886218023355
// MI455X (gfx1250) — matrix operand order repaired and checked by reading, not compiled, not run on hardware
//
#include <hip/hip_runtime.h>
#include <hip/hip_bf16.h>
#include <stdint.h>

// ---------------------------------------------------------------------------
// CDNA5 (gfx1250) fused attention layer:
//   1) qkv = hidden @ W_qkv              (WMMA bf16 GEMM, double-buffered LDS)
//   2) bias + interleaved RoPE -> q/k/v  (elementwise, bf16 staging)
//   3) causal GQA flash attention        (WMMA bf16; K tiles async-pipelined
//                                         via GLOBAL_LOAD_ASYNC_TO_LDS_B128)
//   4) out = attn @ W_dense              (WMMA bf16 GEMM, double-buffered LDS)
//
// Every WMMA fragment is two aligned contiguous 16B LDS reads (ds_load_b128);
// B-operands staged TRANSPOSED (K-major).  LDS rows padded to 80B so 16-lane
// fragment reads walk 16 distinct bank groups (64-bank LDS).
// ---------------------------------------------------------------------------

#define T_TOK     2048
#define HID       4096
#define QKV_OUT   4608      // (32 + 2*2) * 128
#define NQH       32
#define NKVH      2
#define HEAD      128

#define APAD 40             // padded LDS row (bf16 elems), 80B = 16B-aligned
#define KPAD 136            // padded 128-wide bf16 row, 272B

typedef __attribute__((ext_vector_type(16))) __bf16 v16bf;
typedef __attribute__((ext_vector_type(8)))  float  v8f;

// Hardware RNE f32->bf16 (lowers to v_cvt_pk_bf16_f32 on gfx1250).
__device__ __forceinline__ __bf16 f2bf(float f) { return (__bf16)f; }

__device__ __forceinline__ v8f v8f_zero() {
    v8f z;
#pragma unroll
    for (int i = 0; i < 8; ++i) z[i] = 0.0f;
    return z;
}

// Async global->LDS 16B copy (tracked by ASYNCcnt).  lds address = low 32
// bits of a generic pointer into __shared__ ({SHARED_BASE, offset} on AMDGPU).
__device__ __forceinline__ void async_load_lds_b128(void* lds_ptr, const void* g) {
    uint32_t lds_addr = (uint32_t)(uintptr_t)lds_ptr;
    asm volatile("global_load_async_to_lds_b128 %0, %1, off"
                 :: "v"(lds_addr), "v"(g) : "memory");
}

template <int N>
__device__ __forceinline__ void wait_asynccnt() {
#if __has_builtin(__builtin_amdgcn_s_wait_asynccnt)
    __builtin_amdgcn_s_wait_asynccnt(N);
#else
    asm volatile("s_wait_asynccnt %0" :: "i"(N) : "memory");
#endif
}

// 16-bit A-layout fragment from 32 contiguous K elems (caller pre-offsets by
// b1 = lane<16?0:8): lanes 0-15 take K{0..7,16..23}, lanes 16-31 K{8..15,24..31}.
__device__ __forceinline__ v16bf load_afrag(const __bf16* src) {
    v16bf a;
#pragma unroll
    for (int i = 0; i < 8; ++i) { a[i] = src[i]; a[8 + i] = src[16 + i]; }
    return a;
}

// B-layout fragment from K-major storage (caller pre-offsets by lane<16?0:8):
// lanes 0-15 hold K{0..7,16..23} of column N, lanes 16-31 hold K{8..15,24..31}.
__device__ __forceinline__ v16bf load_bfrag(const __bf16* src) {
    v16bf b;
#pragma unroll
    for (int i = 0; i < 8; ++i) { b[i] = src[i]; b[8 + i] = src[16 + i]; }
    return b;
}

// ---------------------------------------------------------------------------
// Kernel 1/4: C[M,N] = A[M,K] * B[K,N], f32 HBM -> bf16 LDS -> WMMA f32.
// Block tile 128x128x32, 8 waves (wave tile 64x32 = 4x2 WMMA accumulators),
// ping-pong LDS double buffering: next tile's global loads overlap WMMAs.
// ---------------------------------------------------------------------------
__global__ __launch_bounds__(256) void gemm_bf16_wmma(
    const float* __restrict__ A, const float* __restrict__ B,
    float* __restrict__ C, int M, int N, int K) {
    __shared__ __bf16 As[2][128][APAD];   // [buf][row][k]  (M-major)
    __shared__ __bf16 Bt[2][128][APAD];   // [buf][col][k]  (N-major, transposed)

    const int tid  = threadIdx.x;
    const int lane = tid & 31;
    const int wave = tid >> 5;
    const int wm   = wave >> 2;          // 0..1  (64-row slab)
    const int wn   = wave & 3;           // 0..3  (32-col slab)
    const int bm0  = blockIdx.y * 128;
    const int bn0  = blockIdx.x * 128;
    const int l15  = lane & 15;
    const int b1   = (lane < 16) ? 0 : 8;    // A-frag K sub-base
    const int kb   = (lane < 16) ? 0 : 8;    // B-frag K sub-base

    v8f acc[4][2];
#pragma unroll
    for (int mi = 0; mi < 4; ++mi)
#pragma unroll
        for (int ni = 0; ni < 2; ++ni) acc[mi][ni] = v8f_zero();

    // Per-thread staging coordinates (coalesced 16-wide f32 global reads).
    const int ar = tid >> 1, ac = (tid & 1) * 16;        // A: 128 rows x 32
    const int br = tid >> 3, bc = (tid & 7) * 16;        // B: 32 rows x 128

    // Prologue: stage tile 0 into buffer 0.
    {
        const float* ag = A + (bm0 + ar) * K + ac;
#pragma unroll
        for (int i = 0; i < 16; ++i) As[0][ar][ac + i] = f2bf(ag[i]);
        const float* bg = B + br * N + bn0 + bc;
#pragma unroll
        for (int i = 0; i < 16; ++i) Bt[0][bc + i][br] = f2bf(bg[i]);
    }

    int buf = 0;
    for (int k0 = 0; k0 < K; k0 += 32) {
        __syncthreads();                       // LDS[buf] ready for everyone
        const bool has_next = (k0 + 32 < K);

        // Issue next tile's global loads early; they drain while WMMAs run.
        float av[16], bv[16];
        if (has_next) {
            const float* ag = A + (bm0 + ar) * K + (k0 + 32) + ac;
#pragma unroll
            for (int i = 0; i < 16; ++i) av[i] = ag[i];
            const float* bg = B + (k0 + 32 + br) * N + bn0 + bc;
#pragma unroll
            for (int i = 0; i < 16; ++i) bv[i] = bg[i];
        }

        v16bf bfrag[2];
#pragma unroll
        for (int ni = 0; ni < 2; ++ni)
            bfrag[ni] = load_bfrag(&Bt[buf][wn * 32 + ni * 16 + l15][kb]);
#pragma unroll
        for (int mi = 0; mi < 4; ++mi) {
            const v16bf afrag = load_afrag(&As[buf][wm * 64 + mi * 16 + l15][b1]);
#pragma unroll
            for (int ni = 0; ni < 2; ++ni)
                acc[mi][ni] = __builtin_amdgcn_wmma_f32_16x16x32_bf16(
                    false, afrag, false, bfrag[ni], (short)0, acc[mi][ni],
                    false, false);
        }

        if (has_next) {                        // convert + store into other buffer
#pragma unroll
            for (int i = 0; i < 16; ++i) As[buf ^ 1][ar][ac + i] = f2bf(av[i]);
#pragma unroll
            for (int i = 0; i < 16; ++i) Bt[buf ^ 1][bc + i][br] = f2bf(bv[i]);
        }
        buf ^= 1;
    }

    // Epilogue with 32-bit address math (all indices < 2^31).
    const int rowb = (lane < 16) ? 0 : 8;
#pragma unroll
    for (int mi = 0; mi < 4; ++mi)
#pragma unroll
        for (int ni = 0; ni < 2; ++ni) {
            float* cp = C + (bm0 + wm * 64 + mi * 16 + rowb) * N
                          + (bn0 + wn * 32 + ni * 16 + l15);
#pragma unroll
            for (int r = 0; r < 8; ++r) cp[r * N] = acc[mi][ni][r];
        }
}

// ---------------------------------------------------------------------------
// Kernel 2/4: bias + interleaved RoPE, scatter to head-major bf16 Q/K/V.
//   e' = e*cos - o*sin ; o' = o*cos + e*sin   for dims < 64 of Q/K heads.
// ---------------------------------------------------------------------------
__global__ __launch_bounds__(128) void biasrope_kernel(
    const float* __restrict__ qkv, const float* __restrict__ bias,
    const float* __restrict__ cosb, const float* __restrict__ sinb,
    __bf16* __restrict__ q_ws, __bf16* __restrict__ k_ws,
    __bf16* __restrict__ v_ws) {
    const int h = blockIdx.x;           // 0..35
    const int t = blockIdx.y;           // 0..2047
    const int d = threadIdx.x;          // 0..127
    const int off = h * HEAD;
    const float* row = qkv + t * QKV_OUT;

    float x = row[off + d] + bias[off + d];
    if (h < NQH + NKVH && d < 64) {     // RoPE region of Q/K heads
        const int i = d >> 1;
        const float c = cosb[t * 64 + i];
        const float s = sinb[t * 64 + i];
        const float e = row[off + 2 * i]     + bias[off + 2 * i];
        const float o = row[off + 2 * i + 1] + bias[off + 2 * i + 1];
        x = (d & 1) ? (o * c + e * s) : (e * c - o * s);
    }
    const __bf16 v = f2bf(x);
    if (h < NQH)
        q_ws[(h * T_TOK + t) * HEAD + d] = v;
    else if (h < NQH + NKVH)
        k_ws[((h - NQH) * T_TOK + t) * HEAD + d] = v;
    else
        v_ws[((h - NQH - NKVH) * T_TOK + t) * HEAD + d] = v;
}

// ---------------------------------------------------------------------------
// Kernel 3/4: causal flash attention.  Block = (head, 128-query tile),
// 8 waves x 16 queries; 32-key tiles; online softmax in f32.
// K tiles flow through a 2-deep async LDS ring (GLOBAL_LOAD_ASYNC_TO_LDS_B128
// + s_wait_asynccnt): tile kt+1 copies while tile kt computes.  V staged
// TRANSPOSED through VGPRs so P*V B-fragments are contiguous b128 loads.
// ---------------------------------------------------------------------------
__global__ __launch_bounds__(256) void attn_kernel(
    const __bf16* __restrict__ q_ws, const __bf16* __restrict__ k_ws,
    const __bf16* __restrict__ v_ws, float* __restrict__ attn_out) {
    __shared__ __bf16 Kt[2][32][KPAD];    // async ring: [buf][key][d]
    __shared__ __bf16 VtT[HEAD][APAD];    // [d][key]  (transposed)
    __shared__ float  Sld[8][16][33];
    __shared__ __bf16 Pld[8][16][APAD];
    __shared__ float  alphaLd[8][16];
    __shared__ float  lLd[8][16];

    const int tid  = threadIdx.x;
    const int lane = tid & 31;
    const int wave = tid >> 5;
    const int h    = blockIdx.x;                 // q head
    const int qb   = blockIdx.y;                 // query tile (128 rows)
    const int kvh  = h >> 4;                     // GQA group of 16
    const int qwb  = qb * 128 + wave * 16;       // wave's first query
    const int l15  = lane & 15;
    const int b1   = (lane < 16) ? 0 : 8;
    const int kb   = (lane < 16) ? 0 : 8;
    const int rowb = (lane < 16) ? 0 : 8;
    const float scale = 0.08838834764831845f;    // 1/sqrt(128)

    // Q fragments for this wave's 16 rows (4 k-steps over d=128), in regs.
    v16bf qf[4];
    {
        const __bf16* qrow = q_ws + (h * T_TOK + qwb + l15) * HEAD;
#pragma unroll
        for (int ks = 0; ks < 4; ++ks) qf[ks] = load_afrag(qrow + ks * 32 + b1);
    }

    v8f o[8];
#pragma unroll
    for (int nt = 0; nt < 8; ++nt) o[nt] = v8f_zero();
    float m = -__builtin_inff();
    float l = 0.0f;

    // Staging coords: 32 keys x 128 dims, 16 elems (32B) per thread.
    const int sr = tid >> 3, sc = (tid & 7) * 16;
    const int ntiles = (qb + 1) * 4;             // causal bound for the block
    const __bf16* kbase_g = k_ws + (kvh * T_TOK + sr) * HEAD + sc;
    const __bf16* vbase_g = v_ws + (kvh * T_TOK + sr) * HEAD + sc;

    // Prologue: async-issue K tile 0 into ring slot 0.
    async_load_lds_b128(&Kt[0][sr][sc],     kbase_g);
    async_load_lds_b128(&Kt[0][sr][sc + 8], kbase_g + 8);

    for (int kt = 0; kt < ntiles; ++kt) {
        const int kbase = kt * 32;
        const int cur   = kt & 1;
        __syncthreads();   // VtT/Pld/Sld consumed; Kt[cur^1] readers done
        {
            // V tile kt: load through VGPRs, store transposed.
            const __bf16* vg = vbase_g + kbase * HEAD;
#pragma unroll
            for (int i = 0; i < 16; ++i) VtT[sc + i][sr] = vg[i];
        }
        if (kt + 1 < ntiles) {
            // Async-issue K tile kt+1 into the other ring slot, then drain to
            // <=2 outstanding: asyncs complete in order, so Kt[cur] is ready.
            const __bf16* kg = kbase_g + (kbase + 32) * HEAD;
            async_load_lds_b128(&Kt[cur ^ 1][sr][sc],     kg);
            async_load_lds_b128(&Kt[cur ^ 1][sr][sc + 8], kg + 8);
            __builtin_prefetch(vbase_g + (kbase + 32) * HEAD, 0, 1);
            wait_asynccnt<2>();
        } else {
            wait_asynccnt<0>();
        }
        __syncthreads();

        // S = Q * K^T  (16 queries x 32 keys)
        v8f sacc[2];
        sacc[0] = v8f_zero(); sacc[1] = v8f_zero();
#pragma unroll
        for (int ks = 0; ks < 4; ++ks) {
#pragma unroll
            for (int ni = 0; ni < 2; ++ni) {
                const v16bf bfrag = load_bfrag(&Kt[cur][ni * 16 + l15][ks * 32 + kb]);
                sacc[ni] = __builtin_amdgcn_wmma_f32_16x16x32_bf16(
                    false, qf[ks], false, bfrag, (short)0, sacc[ni], false, false);
            }
        }
#pragma unroll
        for (int ni = 0; ni < 2; ++ni)
#pragma unroll
            for (int r = 0; r < 8; ++r)
                Sld[wave][rowb + r][ni * 16 + l15] = sacc[ni][r];
        __syncthreads();

        // Online softmax: lanes 0..15 each own one query row.
        if (lane < 16) {
            const int row = lane;
            const int qg  = qwb + row;
            float s[32];
            float mloc = -__builtin_inff();
#pragma unroll
            for (int j = 0; j < 32; ++j) {
                float sv = Sld[wave][row][j] * scale;
                if (kbase + j > qg) sv = -__builtin_inff();   // causal mask
                s[j] = sv;
                mloc = fmaxf(mloc, sv);
            }
            const float mnew  = fmaxf(m, mloc);
            const float alpha = __expf(m - mnew);
            float sum = 0.0f;
#pragma unroll
            for (int j = 0; j < 32; ++j) {
                float p = __expf(s[j] - mnew);
                sum += p;
                Pld[wave][row][j] = f2bf(p);
            }
            l = l * alpha + sum;
            m = mnew;
            alphaLd[wave][row] = alpha;
        }
        __syncthreads();

        // Rescale O by alpha, then O += P * V (8 WMMAs over d).
        float al[8];
#pragma unroll
        for (int r = 0; r < 8; ++r) al[r] = alphaLd[wave][rowb + r];
#pragma unroll
        for (int nt = 0; nt < 8; ++nt)
#pragma unroll
            for (int r = 0; r < 8; ++r) o[nt][r] *= al[r];

        const v16bf pf = load_afrag(&Pld[wave][l15][b1]);
#pragma unroll
        for (int nt = 0; nt < 8; ++nt) {
            const v16bf vfrag = load_bfrag(&VtT[nt * 16 + l15][kb]);
            o[nt] = __builtin_amdgcn_wmma_f32_16x16x32_bf16(
                false, pf, false, vfrag, (short)0, o[nt], false, false);
        }
    }

    // Normalize by l and write attn (T x 4096 f32, head-concatenated).
    if (lane < 16) lLd[wave][lane] = l;
    __syncthreads();
    float inv[8];
#pragma unroll
    for (int r = 0; r < 8; ++r) inv[r] = 1.0f / lLd[wave][rowb + r];
#pragma unroll
    for (int nt = 0; nt < 8; ++nt) {
        float* op = attn_out + (qwb + rowb) * (NQH * HEAD) + h * HEAD + nt * 16 + l15;
#pragma unroll
        for (int r = 0; r < 8; ++r) op[r * (NQH * HEAD)] = o[nt][r] * inv[r];
    }
}

// ---------------------------------------------------------------------------
// Host-side orchestration.
// d_in: 0=hidden(2048x4096) 1=cos(2048x64) 2=sin(2048x64)
//       3=W_qkv(4096x4608) 4=b_qkv(4608) 5=W_dense(4096x4096); all f32.
// ---------------------------------------------------------------------------
extern "C" void kernel_launch(void* const* d_in, const int* in_sizes, int n_in,
                              void* d_out, int out_size, void* d_ws, size_t ws_size,
                              hipStream_t stream) {
    (void)in_sizes; (void)n_in; (void)out_size; (void)ws_size;
    const float* hidden  = (const float*)d_in[0];
    const float* cosb    = (const float*)d_in[1];
    const float* sinb    = (const float*)d_in[2];
    const float* W_qkv   = (const float*)d_in[3];
    const float* b_qkv   = (const float*)d_in[4];
    const float* W_dense = (const float*)d_in[5];
    float* out = (float*)d_out;

    // Workspace carve-up (~90 MB).
    char* ws = (char*)d_ws;
    float*  qkv_ws  = (float*)ws;                             // 2048*4608 f32
    ws += (size_t)T_TOK * QKV_OUT * sizeof(float);
    __bf16* q_ws    = (__bf16*)ws;                            // 32*2048*128 bf16
    ws += (size_t)NQH * T_TOK * HEAD * sizeof(__bf16);
    __bf16* k_ws    = (__bf16*)ws;                            // 2*2048*128 bf16
    ws += (size_t)NKVH * T_TOK * HEAD * sizeof(__bf16);
    __bf16* v_ws    = (__bf16*)ws;
    ws += (size_t)NKVH * T_TOK * HEAD * sizeof(__bf16);
    float*  attn_ws = (float*)ws;                             // 2048*4096 f32

    // 1) qkv = hidden @ W_qkv
    gemm_bf16_wmma<<<dim3(QKV_OUT / 128, T_TOK / 128), 256, 0, stream>>>(
        hidden, W_qkv, qkv_ws, T_TOK, QKV_OUT, HID);

    // 2) bias + RoPE -> bf16 q/k/v (head-major)
    biasrope_kernel<<<dim3(NQH + 2 * NKVH, T_TOK), 128, 0, stream>>>(
        qkv_ws, b_qkv, cosb, sinb, q_ws, k_ws, v_ws);

    // 3) causal GQA flash attention
    attn_kernel<<<dim3(NQH, T_TOK / 128), 256, 0, stream>>>(
        q_ws, k_ws, v_ws, attn_ws);

    // 4) out = attn @ W_dense
    gemm_bf16_wmma<<<dim3(HID / 128, T_TOK / 128), 256, 0, stream>>>(
        attn_ws, W_dense, out, T_TOK, HID, NQH * HEAD);
}
